// GptOssAttention_6493990551766
// MI455X (gfx1250) — matrix operand order repaired and checked by reading, not compiled, not run on hardware
//
#include <hip/hip_runtime.h>
#include <hip/hip_bf16.h>
#include <math.h>

// ---------------------------------------------------------------------------
// CDNA5 (gfx1250) wave32 WMMA attention pipeline, v4.
// - all matmuls: v_wmma_f32_16x16x32_bf16 (fp32 accumulate)
// - A-fragment LDS traffic: 2 x ds_load_b128 / ds_store_b128 per fragment; B fragments gathered per u16
// - global -> LDS staging: global_load_async_to_lds_b128 with double-buffered
//   LDS tiles, synchronized with s_wait_asynccnt (per-wave counter, in-order
//   completion => waiting to <= N outstanding retires the previous tile).
// - sliding-window mask as a single unsigned compare per element.
// ---------------------------------------------------------------------------

typedef __attribute__((ext_vector_type(16))) __bf16 v16bf;
typedef __attribute__((ext_vector_type(8)))  float  v8f;

union FragBF { v16bf v; unsigned short u[16]; uint4 q[2]; };

__device__ __forceinline__ unsigned short f2bf(float f) {
  unsigned int u = __float_as_uint(f);
  u += 0x7FFFu + ((u >> 16) & 1u);            // round-to-nearest-even
  return (unsigned short)(u >> 16);
}

// Async 16B copy global -> LDS (VDST = LDS byte address; generic LDS pointers
// carry the workgroup-relative offset in their low 32 bits, ISA 10.2).
__device__ __forceinline__ void async_b128(const void* gaddr, void* lds) {
  asm volatile("global_load_async_to_lds_b128 %0, %1, off"
               :: "v"((unsigned int)(unsigned long long)(size_t)lds),
                  "v"((unsigned long long)(size_t)gaddr)
               : "memory");
}
template<int N>
__device__ __forceinline__ void wait_asynccnt() {
  asm volatile("s_wait_asynccnt %0" :: "n"(N) : "memory");
}

// A-layout fragment: runs at [p .. p+7] and [p+16 .. p+23] (p = base + hi*8)
__device__ __forceinline__ v16bf ld_frag_a(const unsigned short* p) {
  FragBF f;
  f.q[0] = *reinterpret_cast<const uint4*>(p);
  f.q[1] = *reinterpret_cast<const uint4*>(p + 16);
  return f.v;
}
// B-layout fragment: 16 contiguous elements per lane
__device__ __forceinline__ v16bf ld_frag_b(const unsigned short* p) {
  FragBF f;
  f.q[0] = *reinterpret_cast<const uint4*>(p);
  f.q[1] = *reinterpret_cast<const uint4*>(p + 8);
  return f.v;
}

// ---------------------------------------------------------------------------
// Elementwise fp32 -> bf16 (x4); run once per tensor.
// ---------------------------------------------------------------------------
__global__ void cvt_bf16(const float* __restrict__ in,
                         unsigned short* __restrict__ out, int n4) {
  const int i = blockIdx.x * blockDim.x + threadIdx.x;
  if (i >= n4) return;
  const float4 f = reinterpret_cast<const float4*>(in)[i];
  const unsigned int lo = (unsigned)f2bf(f.x) | ((unsigned)f2bf(f.y) << 16);
  const unsigned int hi = (unsigned)f2bf(f.z) | ((unsigned)f2bf(f.w) << 16);
  reinterpret_cast<uint2*>(out)[i] = make_uint2(lo, hi);
}

// ---------------------------------------------------------------------------
// Tiled bf16 GEMM: C[M,N] = A[M,K] * B[K,N], fp32 accumulate.
// Block tile 128x128, K-tile 64, 8 waves; wave tile 64x32 (4x2 fragments).
// Double-buffered async LDS staging: 8 async b128 issues / thread / tile.
// ---------------------------------------------------------------------------
template<bool OUT_BF16>
__global__ __launch_bounds__(256) void wmma_gemm(
    const unsigned short* __restrict__ A, const unsigned short* __restrict__ B,
    void* __restrict__ Cp, int M, int N, int K) {
  constexpr int BM = 128, BN = 128, BK = 64;
  constexpr int ASTR = BK + 8;   // 72 ushorts = 144 B rows (16B-aligned)
  constexpr int BSTR = BN + 8;   // 136 ushorts = 272 B rows (16B-aligned)
  __shared__ __align__(16) unsigned short As[2][BM * ASTR];
  __shared__ __align__(16) unsigned short Bs[2][BK * BSTR];

  const int tid  = threadIdx.x;
  const int lane = tid & 31;
  const int wave = tid >> 5;
  const int wm   = (wave >> 2) * 64;
  const int wn   = (wave & 3) * 32;
  const int bm   = blockIdx.y * BM;
  const int bn   = blockIdx.x * BN;
  const int mrow = lane & 15;
  const int hi   = lane >> 4;

  // staging coordinates (16B granules)
  const int ac  = (tid & 7) * 8;
  const int ar0 = tid >> 3;                    // 0..31
  const int bc  = (tid & 15) * 8;
  const int br0 = tid >> 4;                    // 0..15

  auto issue_tile = [&](int buf, int kb) {
    #pragma unroll
    for (int i = 0; i < 4; ++i) {
      const int row = ar0 + i * 32;
      async_b128(A + (size_t)(bm + row) * K + kb + ac,
                 &As[buf][row * ASTR + ac]);
    }
    #pragma unroll
    for (int i = 0; i < 4; ++i) {
      const int row = br0 + i * 16;
      async_b128(B + (size_t)(kb + row) * N + bn + bc,
                 &Bs[buf][row * BSTR + bc]);
    }
  };

  v8f acc[4][2];
  for (int i = 0; i < 4; ++i)
    for (int j = 0; j < 2; ++j) acc[i][j] = (v8f)0.0f;

  issue_tile(0, 0);
  int buf = 0;
  for (int kb = 0; kb < K; kb += BK) {
    const bool has_next = (kb + BK) < K;
    if (has_next) issue_tile(buf ^ 1, kb + BK);   // overlaps with this tile
    if (has_next) wait_asynccnt<8>();             // retire current tile only
    else          wait_asynccnt<0>();
    __syncthreads();

    const unsigned short* Asb = &As[buf][0];
    const unsigned short* Bsb = &Bs[buf][0];
    #pragma unroll
    for (int cc = 0; cc < 2; ++cc) {              // two K=32 WMMA chunks
      v16bf af[4], bfr[2];
      #pragma unroll
      for (int mi = 0; mi < 4; ++mi)
        af[mi] = ld_frag_a(&Asb[(wm + mi * 16 + mrow) * ASTR + cc * 32 + hi * 8]);
      #pragma unroll
      for (int ni = 0; ni < 2; ++ni) {
        FragBF fb;
        #pragma unroll
        for (int e = 0; e < 16; ++e)
          fb.u[e] = Bsb[(cc * 32 + hi * 8 + e + (e & 8)) * BSTR + wn + ni * 16 + mrow];
        bfr[ni] = fb.v;
      }
      #pragma unroll
      for (int mi = 0; mi < 4; ++mi)
        #pragma unroll
        for (int ni = 0; ni < 2; ++ni)
          acc[mi][ni] = __builtin_amdgcn_wmma_f32_16x16x32_bf16(
              false, af[mi], false, bfr[ni], (short)0, acc[mi][ni],
              false, false);
    }
    __syncthreads();   // all waves done reading buf before next overwrite
    buf ^= 1;
  }

  #pragma unroll
  for (int mi = 0; mi < 4; ++mi)
    #pragma unroll
    for (int ni = 0; ni < 2; ++ni)
      #pragma unroll
      for (int r = 0; r < 8; ++r) {
        const int m = bm + wm + mi * 16 + r + 8 * hi;
        const int n = bn + wn + ni * 16 + (lane & 15);
        if (OUT_BF16)
          ((unsigned short*)Cp)[(size_t)m * N + n] = f2bf(acc[mi][ni][r]);
        else
          ((float*)Cp)[(size_t)m * N + n] = acc[mi][ni][r];
      }
}

// ---------------------------------------------------------------------------
// RoPE (rotate-half), fp32 math, bf16 out; query scale folded in.
// ---------------------------------------------------------------------------
__global__ void rope_kernel(const float* __restrict__ in,
                            unsigned short* __restrict__ out,
                            int nh, float scale, int Slen) {
  const int idx = blockIdx.x * blockDim.x + threadIdx.x;
  const int i   = idx & 63;
  const int t   = idx >> 6;
  const int hd  = t % nh;
  const int s   = t / nh;
  if (s >= Slen) return;
  const size_t base = ((size_t)s * nh + hd) * 128;
  const float invf = powf(10000.0f, -(float)i * (1.0f / 64.0f));
  const float f  = (float)s * invf;
  const float cs = cosf(f), sn = sinf(f);
  const float x1 = in[base + i], x2 = in[base + i + 64];
  out[base + i]      = f2bf((x1 * cs - x2 * sn) * scale);
  out[base + i + 64] = f2bf((x2 * cs + x1 * sn) * scale);
}

// ---------------------------------------------------------------------------
// Flash attention, sliding window + sink; 1 wave per (16-query tile, head).
// S^T = K * Q^T form (K as A-fragments in table k order, per-lane scalar softmax stats).
// K/V tiles double-buffered via async global->LDS (32 issues / wave / tile).
// Sink folded into softmax init (m0 = sink, l0 = 1).
// ---------------------------------------------------------------------------
__global__ __launch_bounds__(32) void attn_kernel(
    const unsigned short* __restrict__ Qb,   // (S,32,128) bf16, pre-scaled
    const unsigned short* __restrict__ Kb,   // (S, 8,128) bf16
    const unsigned short* __restrict__ Vb,   // (S, 8,128) bf16
    const float* __restrict__ sinks,
    unsigned short* __restrict__ Ob) {       // (S,32,128) bf16
  constexpr int S_ = 2048, NQh = 32, NKVh = 8, H_ = 128, WIN = 1024;
  constexpr int KSTR = H_ + 8;   // 136 ushorts = 272 B rows
  constexpr int PSTR = 40;       // 80 B rows
  __shared__ __align__(16) unsigned short Ks[2][32 * KSTR];
  __shared__ __align__(16) unsigned short Vs[2][32 * KSTR];
  __shared__ __align__(16) unsigned short Ps[16 * PSTR];

  const int lane = threadIdx.x;
  const int qi0  = blockIdx.x * 16;
  const int u    = blockIdx.y;
  const int kvh  = u >> 2;
  const int mrow = lane & 15;
  const int hi   = lane >> 4;

  // Q as B-fragments (lane = query, e = h within chunk), gathered once.
  FragBF qbf[4];
  #pragma unroll
  for (int c = 0; c < 4; ++c)
    #pragma unroll
    for (int e = 0; e < 16; ++e)
      qbf[c].u[e] = Qb[(((size_t)(qi0 + mrow) * NQh) + u) * H_ + c * 32 + hi * 8 + e + (e & 8)];

  // one 256B K row + one 256B V row per lane, async (32 issues per tile)
  auto issue_kv = [&](int buf, int kb) {
    const int krow = kb + lane;
    if (krow < S_) {
      const unsigned short* kg = Kb + (((size_t)krow * NKVh) + kvh) * H_;
      const unsigned short* vg = Vb + (((size_t)krow * NKVh) + kvh) * H_;
      #pragma unroll
      for (int j = 0; j < 16; ++j)
        async_b128(kg + j * 8, &Ks[buf][lane * KSTR + j * 8]);
      #pragma unroll
      for (int j = 0; j < 16; ++j)
        async_b128(vg + j * 8, &Vs[buf][lane * KSTR + j * 8]);
    } else {
      const uint4 z = make_uint4(0u, 0u, 0u, 0u);
      #pragma unroll
      for (int j = 0; j < 16; ++j) {
        *(uint4*)&Ks[buf][lane * KSTR + j * 8] = z;
        *(uint4*)&Vs[buf][lane * KSTR + j * 8] = z;
      }
    }
  };

  v8f oacc[8];
  for (int i = 0; i < 8; ++i) oacc[i] = (v8f)0.0f;
  float mrun = sinks[u];         // per-lane stats for query = lane&15
  float lrun = 1.0f;

  const int kb0 = (qi0 > (WIN - 1)) ? ((qi0 - (WIN - 1)) & ~31) : 0;
  issue_kv(0, kb0);
  int buf = 0;
  for (int kb = kb0; kb <= qi0 + 15; kb += 32) {
    const bool has_next = (kb + 32) <= (qi0 + 15);
    if (has_next) issue_kv(buf ^ 1, kb + 32);
    if (has_next) wait_asynccnt<32>();
    else          wait_asynccnt<0>();
    __syncthreads();   // S_NOP (single wave) but orders DS/ASYNC traffic

    const unsigned short* Ksb = &Ks[buf][0];
    const unsigned short* Vsb = &Vs[buf][0];

    // S^T tiles: M = key (two tiles of 16), N = query, contraction over H.
    v8f sct[2];
    sct[0] = (v8f)0.0f; sct[1] = (v8f)0.0f;
    #pragma unroll
    for (int t = 0; t < 2; ++t)
      #pragma unroll
      for (int c = 0; c < 4; ++c) {
        const v16bf kf =
            ld_frag_a(&Ksb[(t * 16 + mrow) * KSTR + c * 32 + hi * 8]);
        sct[t] = __builtin_amdgcn_wmma_f32_16x16x32_bf16(
            false, kf, false, qbf[c].v, (short)0, sct[t], false, false);
      }

    // mask + online softmax (lane owns query q = qi0 + (lane&15)).
    // window predicate folds to ONE unsigned compare: 0 <= q-key < WIN.
    const int q  = qi0 + (lane & 15);
    const int d0 = q - (kb + 8 * hi);          // q - key = d0 - t*16 - r
    float pv[2][8];
    float locmax = -3.0e38f;
    #pragma unroll
    for (int t = 0; t < 2; ++t)
      #pragma unroll
      for (int r = 0; r < 8; ++r) {
        float s = sct[t][r];
        if ((unsigned)(d0 - t * 16 - r) >= (unsigned)WIN) s = -1e30f;
        pv[t][r] = s;
        locmax = fmaxf(locmax, s);
      }
    locmax = fmaxf(locmax, __shfl_xor(locmax, 16, 32));  // join key halves
    const float mnew  = fmaxf(mrun, locmax);
    const float alpha = __expf(mrun - mnew);
    float psum = 0.0f;
    #pragma unroll
    for (int t = 0; t < 2; ++t)
      #pragma unroll
      for (int r = 0; r < 8; ++r) {
        const float e = __expf(pv[t][r] - mnew);
        pv[t][r] = e;
        psum += e;
      }
    psum += __shfl_xor(psum, 16, 32);
    lrun = lrun * alpha + psum;
    mrun = mnew;

    // rescale O accumulators (row q = r + 8*hi; stats live in lane q)
    #pragma unroll
    for (int r = 0; r < 8; ++r) {
      const float ar = __shfl(alpha, r + 8 * hi, 32);
      #pragma unroll
      for (int ht = 0; ht < 8; ++ht) oacc[ht][r] *= ar;
    }

    // pack P^T -> Ps[q][key]: keys t*16 + hi*8 + (0..7) consecutive in r
    #pragma unroll
    for (int t = 0; t < 2; ++t) {
      uint4 w;
      w.x = (unsigned)f2bf(pv[t][0]) | ((unsigned)f2bf(pv[t][1]) << 16);
      w.y = (unsigned)f2bf(pv[t][2]) | ((unsigned)f2bf(pv[t][3]) << 16);
      w.z = (unsigned)f2bf(pv[t][4]) | ((unsigned)f2bf(pv[t][5]) << 16);
      w.w = (unsigned)f2bf(pv[t][6]) | ((unsigned)f2bf(pv[t][7]) << 16);
      *(uint4*)&Ps[(lane & 15) * PSTR + t * 16 + hi * 8] = w;
    }
    __syncthreads();

    // O += P(16x32) * V(32x128)
    const v16bf pa = ld_frag_a(&Ps[mrow * PSTR + hi * 8]);
    #pragma unroll
    for (int ht = 0; ht < 8; ++ht) {
      FragBF vfb;
      #pragma unroll
      for (int e = 0; e < 16; ++e)
        vfb.u[e] = Vsb[(hi * 8 + e + (e & 8)) * KSTR + ht * 16 + mrow];
      const v16bf vb = vfb.v;
      oacc[ht] = __builtin_amdgcn_wmma_f32_16x16x32_bf16(
          false, pa, false, vb, (short)0, oacc[ht], false, false);
    }
    __syncthreads();
    buf ^= 1;
  }

  // normalize + store bf16 O (S,32,128)
  #pragma unroll
  for (int r = 0; r < 8; ++r) {
    const float inv = 1.0f / __shfl(lrun, r + 8 * hi, 32);
    const int m = qi0 + r + 8 * hi;
    #pragma unroll
    for (int ht = 0; ht < 8; ++ht) {
      const int col = ht * 16 + (lane & 15);
      Ob[(((size_t)m * NQh) + u) * H_ + col] = f2bf(oacc[ht][r] * inv);
    }
  }
}

// ---------------------------------------------------------------------------
// Host orchestration. Workspace 128 MB with stream-ordered aliasing:
//   Hb 16M | Wkb 8M | Wvb 8M | Wqb 32M (-> Obf after Q GEMM)
//   | Qf 32M (-> Wob after RoPE-Q) | Kf 8M | Qbf 16M | Kbf 4M | Vbf 4M
// ---------------------------------------------------------------------------
extern "C" void kernel_launch(void* const* d_in, const int* in_sizes, int n_in,
                              void* d_out, int out_size, void* d_ws, size_t ws_size,
                              hipStream_t stream) {
  (void)in_sizes; (void)n_in; (void)out_size; (void)ws_size;
  const float* hidden = (const float*)d_in[0];
  const float* Wq     = (const float*)d_in[1];
  const float* Wk     = (const float*)d_in[2];
  const float* Wv     = (const float*)d_in[3];
  const float* Wo     = (const float*)d_in[4];
  const float* sinks  = (const float*)d_in[5];
  // d_in[6] (attention_mask) = sliding-window causal, computed analytically.

  constexpr int S = 2048, M = 4096, NQ = 32, NKV = 8, H = 128;

  char* ws = (char*)d_ws;
  unsigned short* Hb  = (unsigned short*)(ws + 0);           // 16 MB
  unsigned short* Wkb = (unsigned short*)(ws + 16777216);    //  8 MB
  unsigned short* Wvb = (unsigned short*)(ws + 25165824);    //  8 MB
  unsigned short* Wqb = (unsigned short*)(ws + 33554432);    // 32 MB
  unsigned short* Obf = (unsigned short*)(ws + 33554432);    // alias (16 MB)
  float*          Qf  = (float*)(ws + 67108864);             // 32 MB
  unsigned short* Wob = (unsigned short*)(ws + 67108864);    // alias (32 MB)
  float*          Kf  = (float*)(ws + 100663296);            //  8 MB
  unsigned short* Qbf = (unsigned short*)(ws + 109051904);   // 16 MB
  unsigned short* Kbf = (unsigned short*)(ws + 125829120);   //  4 MB
  unsigned short* Vbf = (unsigned short*)(ws + 130023424);   //  4 MB

  // one-time bf16 conversions (data reused 16x by GEMM row-blocks)
  cvt_bf16<<<(S * M / 4 + 255) / 256, 256, 0, stream>>>(hidden, Hb, S * M / 4);
  cvt_bf16<<<(M * NQ * H / 4 + 255) / 256, 256, 0, stream>>>(Wq, Wqb, M * NQ * H / 4);
  cvt_bf16<<<(M * NKV * H / 4 + 255) / 256, 256, 0, stream>>>(Wk, Wkb, M * NKV * H / 4);
  cvt_bf16<<<(M * NKV * H / 4 + 255) / 256, 256, 0, stream>>>(Wv, Wvb, M * NKV * H / 4);

  const dim3 blk(256);
  wmma_gemm<false><<<dim3((NQ * H) / 128, S / 128), blk, 0, stream>>>(
      Hb, Wqb, Qf, S, NQ * H, M);                       // Wqb dead after this
  wmma_gemm<false><<<dim3((NKV * H) / 128, S / 128), blk, 0, stream>>>(
      Hb, Wkb, Kf, S, NKV * H, M);
  wmma_gemm<true><<<dim3((NKV * H) / 128, S / 128), blk, 0, stream>>>(
      Hb, Wvb, Vbf, S, NKV * H, M);

  const float qscale = 1.0f / sqrtf((float)H);
  rope_kernel<<<(S * NQ * 64) / 256, 256, 0, stream>>>(Qf, Qbf, NQ, qscale, S);
  rope_kernel<<<(S * NKV * 64) / 256, 256, 0, stream>>>(Kf, Kbf, NKV, 1.0f, S);

  // Qf consumed -> its region now holds bf16 Wo
  cvt_bf16<<<(NQ * H * M / 4 + 255) / 256, 256, 0, stream>>>(Wo, Wob, NQ * H * M / 4);

  attn_kernel<<<dim3(S / 16, NQ), dim3(32), 0, stream>>>(Qbf, Kbf, Vbf, sinks, Obf);

  wmma_gemm<false><<<dim3(M / 128, S / 128), blk, 0, stream>>>(
      Obf, Wob, (float*)d_out, S, M, NQ * H);
}
